// MultiScaleAttention_49624052138051
// MI455X (gfx1250) — matrix operand order repaired and checked by reading, not compiled, not run on hardware
//
#include <hip/hip_runtime.h>
#include <stdint.h>

typedef unsigned short u16;
typedef __attribute__((ext_vector_type(16))) __bf16 v16bf;
typedef __attribute__((ext_vector_type(16))) unsigned short v16u;
typedef __attribute__((ext_vector_type(8))) float v8f;
typedef __attribute__((ext_vector_type(4))) unsigned int v4u;
typedef __attribute__((ext_vector_type(8))) unsigned int v8u;

#define BATCH 8
#define NTOK  4096
#define CDIM  512
#define NH    8
#define HD    64
#define WGRID 64

// ---------- helpers ----------

__device__ __forceinline__ u16 f2bf(float f) {
  union { float f; unsigned u; } x; x.f = f;
  unsigned r = (x.u + 0x7FFFu + ((x.u >> 16) & 1u)) >> 16;
  return (u16)r;
}

__device__ __forceinline__ v16bf pack_frag(const u16 t[16]) {
  v16u u;
#pragma unroll
  for (int i = 0; i < 16; ++i) u[i] = t[i];
  return __builtin_bit_cast(v16bf, u);
}

// A-matrix fragment (16x32 bf16), contiguous k order (k = 16h + s), matched by the B loaders:
// lanes 0-15: row=lane, VGPR v holds K = v*2 (+16 for lanes 16-31).
__device__ __forceinline__ v16bf load_a_frag(const u16* rowptr, int k0) {
  const int hi = (threadIdx.x >> 4) & 1;
  u16 t[16];
#pragma unroll
  for (int v = 0; v < 8; ++v) {
    int k = k0 + (hi << 4) + (v << 1);
    unsigned pr;
    __builtin_memcpy(&pr, rowptr + k, 4);
    t[2 * v]     = (u16)(pr & 0xFFFFu);
    t[2 * v + 1] = (u16)(pr >> 16);
  }
  return pack_frag(t);
}

// B fragment, memory holds B transposed: mem is N x K row-major,
// B_wmma[k][n] = mem[n*ld + k].  (Q*K^T: mem = K rows; GEMM: weight rows)
__device__ __forceinline__ v16bf load_bT_frag(const u16* base, int ld) {
  const int lane = threadIdx.x & 31;
  const int n = lane & 15, hi = lane >> 4;
  u16 t[16];
#pragma unroll
  for (int v = 0; v < 8; ++v) {
    int k = (v << 1) + (hi << 4);
    unsigned pr;
    __builtin_memcpy(&pr, base + n * ld + k, 4);
    t[2 * v]     = (u16)(pr & 0xFFFFu);
    t[2 * v + 1] = (u16)(pr >> 16);
  }
  return pack_frag(t);
}

// B fragment, memory holds B directly: mem is K x N row-major,
// B_wmma[k][n] = mem[k*ld + n].  (P*V: mem = V rows)
__device__ __forceinline__ v16bf load_bN_frag(const u16* base, int ld) {
  const int lane = threadIdx.x & 31;
  const int n = lane & 15, hi = lane >> 4;
  u16 t[16];
#pragma unroll
  for (int v = 0; v < 8; ++v) {
    int k = (v << 1) + (hi << 4);
    t[2 * v]     = base[k * ld + n];
    t[2 * v + 1] = base[(k + 1) * ld + n];
  }
  return pack_frag(t);
}

__device__ __forceinline__ v8f wmma_bf16(v16bf a, v16bf b, v8f c) {
  return __builtin_amdgcn_wmma_f32_16x16x32_bf16(false, a, false, b, (short)0, c,
                                                 false, false);
}

// ---------- Tensor Data Mover: 2D tile (Global -> LDS), bf16 elements -------
// D# per CDNA5 ISA. workgroup_mask=0 (no cluster).
// pad codes: LDS row padding applied by the TDM itself for bank spreading.
// Every descriptor word is pinned to SGPRs via readfirstlane so the "s"
// constrained tuples always allocate as s[a:a+3] / s[b:b+7] (VADDR0/1 of the
// VIMAGE-encoded TDM op must be scalar registers).
__device__ __forceinline__ void tdm_load_2d(
    unsigned lds_off, const void* g,
    unsigned tensor_d0, unsigned tensor_d1, unsigned stride_elems,
    unsigned tile_d0, unsigned tile_d1,
    unsigned pad_int_code, unsigned pad_amt_code) {
  unsigned long long ga = (unsigned long long)(uintptr_t)g;
  unsigned w0 = 1u;                                        // count=1, user D#
  unsigned w1 = lds_off;                                   // lds_addr (bytes)
  unsigned w2 = (unsigned)ga;                              // global_addr[31:0]
  unsigned w3 = (unsigned)((ga >> 32) & 0x1FFFFFFull) |    // global_addr[56:32]
                (2u << 30);                                // type=2 ("image")
  unsigned u0 = (1u << 16)                 // data_size = 1 (2 bytes)
              | (1u << 20)                 // pad_enable
              | (pad_int_code << 22)
              | (pad_amt_code << 25);
  unsigned u1 = (tensor_d0 & 0xFFFFu) << 16;
  unsigned u2 = ((tensor_d0 >> 16) & 0xFFFFu) | ((tensor_d1 & 0xFFFFu) << 16);
  unsigned u3 = ((tensor_d1 >> 16) & 0xFFFFu) | (tile_d0 << 16);
  unsigned u4 = (tile_d1 & 0xFFFFu);
  unsigned u5 = stride_elems;
  v4u g0;
  g0[0] = (unsigned)__builtin_amdgcn_readfirstlane((int)w0);
  g0[1] = (unsigned)__builtin_amdgcn_readfirstlane((int)w1);
  g0[2] = (unsigned)__builtin_amdgcn_readfirstlane((int)w2);
  g0[3] = (unsigned)__builtin_amdgcn_readfirstlane((int)w3);
  v8u g1;
  g1[0] = (unsigned)__builtin_amdgcn_readfirstlane((int)u0);
  g1[1] = (unsigned)__builtin_amdgcn_readfirstlane((int)u1);
  g1[2] = (unsigned)__builtin_amdgcn_readfirstlane((int)u2);
  g1[3] = (unsigned)__builtin_amdgcn_readfirstlane((int)u3);
  g1[4] = (unsigned)__builtin_amdgcn_readfirstlane((int)u4);
  g1[5] = (unsigned)__builtin_amdgcn_readfirstlane((int)u5);
  g1[6] = 0u;
  g1[7] = 0u;
  asm volatile("tensor_load_to_lds %0, %1" :: "s"(g0), "s"(g1) : "memory");
}

__device__ __forceinline__ unsigned lds_addr_of(const void* p) {
  return (unsigned)(uintptr_t)p;
}

// ---------- fp32 -> bf16 staging ----------

__global__ __launch_bounds__(256) void cvt_bf16(const float* __restrict__ src,
                                                u16* __restrict__ dst, int n) {
  const int stride = gridDim.x * blockDim.x;
  for (int i = blockIdx.x * blockDim.x + threadIdx.x; i < n; i += stride) {
    __builtin_prefetch(src + i + 2 * stride, 0, 1);   // global_prefetch_b8
    dst[i] = f2bf(src[i]);
  }
}

// ---------- WMMA GEMM:  C[m,n] = sum_k A[m,k] * B[n,k]  (A MxK, B NxK, bf16) --
// block = 8 waves, tile 128(M) x 64(N); wave computes 16x64 -> 4 WMMAs/K-step.
// Double-buffered: tile i+1 (TDM for B, b128 loads for A) is staged while the
// WMMAs consume tile i; one barrier + tensorcnt wait per K-step.

__global__ __launch_bounds__(256) void gemm_bf16(
    const u16* __restrict__ A, const u16* __restrict__ Bm, int M, int N, int K,
    float* __restrict__ outf, const float* __restrict__ bias,
    u16* __restrict__ o1, u16* __restrict__ o2, int mode, float scale) {
  __shared__ u16 Bs[2][64 * 36];    // TDM-padded stride 36 (72B rows)
  __shared__ u16 As[2][128 * 48];
  const int tid = threadIdx.x;
  const int w = tid >> 5;
  const int lane = tid & 31;
  const int hi = lane >> 4, col = lane & 15;
  const int m0 = blockIdx.x * 128;
  const int n0 = blockIdx.y * 64;
  v8f acc[4];
#pragma unroll
  for (int nt = 0; nt < 4; ++nt) acc[nt] = (v8f){0.f,0.f,0.f,0.f,0.f,0.f,0.f,0.f};

  auto stage = [&](int step, int buf) {
    if (tid < 32) {
      // async DMA: 64 (N rows) x 32 (K) bf16 weight tile, padded to 72B rows
      tdm_load_2d(lds_addr_of(&Bs[buf][0]), Bm + (size_t)n0 * K + step * 32,
                  (unsigned)K, (unsigned)N, (unsigned)K, 32u, 64u, 3u, 1u);
    }
    int r = tid >> 1, c0 = (tid & 1) << 4;   // A tile: 128x32, 16 elems/thread
    const uint4* g = (const uint4*)(A + (size_t)(m0 + r) * K + step * 32 + c0);
    *(uint4*)&As[buf][r * 48 + c0]     = g[0];
    *(uint4*)&As[buf][r * 48 + c0 + 8] = g[1];
  };

  const int nsteps = K >> 5;
  stage(0, 0);
  if (tid < 32) __builtin_amdgcn_s_wait_tensorcnt(0);
  __syncthreads();
  for (int i = 0; i < nsteps; ++i) {
    const int cur = i & 1;
    if (i + 1 < nsteps) stage(i + 1, cur ^ 1);
    v16bf a = load_a_frag(&As[cur][(w * 16 + col) * 48], 0);
    acc[0] = wmma_bf16(a, load_bT_frag(&Bs[cur][0 * 16 * 36], 36), acc[0]);
    acc[1] = wmma_bf16(a, load_bT_frag(&Bs[cur][1 * 16 * 36], 36), acc[1]);
    acc[2] = wmma_bf16(a, load_bT_frag(&Bs[cur][2 * 16 * 36], 36), acc[2]);
    acc[3] = wmma_bf16(a, load_bT_frag(&Bs[cur][3 * 16 * 36], 36), acc[3]);
    if (tid < 32) __builtin_amdgcn_s_wait_tensorcnt(0);
    __syncthreads();
  }

#pragma unroll
  for (int nt = 0; nt < 4; ++nt) {
#pragma unroll
    for (int e = 0; e < 8; ++e) {
      int gm = m0 + w * 16 + e + (hi << 3);
      int gn = n0 + nt * 16 + col;
      float r = acc[nt][e];
      if (mode == 0) {
        outf[(size_t)gm * N + gn] = r + bias[gn];
      } else if (mode == 1) {
        int bb = gm >> 12, t = gm & 4095;
        int h = gn >> 6, d = gn & 63;
        o1[(((size_t)bb * NH + h) * NTOK + t) * HD + d] = f2bf(r * scale);
      } else {
        int bb = gm >> 12, t = gm & 4095;
        int c = gn;
        u16* dst = o1;
        if (c >= CDIM) { c -= CDIM; dst = o2; }
        int h = c >> 6, d = c & 63;
        dst[(((size_t)bb * NH + h) * NTOK + t) * HD + d] = f2bf(r);
      }
    }
  }
}

// ---------- shared flash-attention step: softmax update + P*V accumulate -----

__device__ __forceinline__ void softmax_pv_step(
    v8f& s0, v8f& s1, float* m_r, float* l_r,
    v8f& o0, v8f& o1, v8f& o2, v8f& o3,
    u16* psw /* wave-private 16x36 LDS */, const u16* vsb /* 32x72 LDS */) {
  const int lane = threadIdx.x & 31;
  const int hi = lane >> 4, col = lane & 15;
#pragma unroll
  for (int e = 0; e < 8; ++e) {
    float mx = fmaxf(s0[e], s1[e]);
#pragma unroll
    for (int off = 1; off < 16; off <<= 1) mx = fmaxf(mx, __shfl_xor(mx, off, 32));
    float mnew = fmaxf(m_r[e], mx);
    float p0 = __expf(s0[e] - mnew);
    float p1 = __expf(s1[e] - mnew);
    float rs = p0 + p1;
#pragma unroll
    for (int off = 1; off < 16; off <<= 1) rs += __shfl_xor(rs, off, 32);
    float al = __expf(m_r[e] - mnew);
    l_r[e] = l_r[e] * al + rs;
    m_r[e] = mnew;
    o0[e] *= al; o1[e] *= al; o2[e] *= al; o3[e] *= al;
    int row = e + (hi << 3);
    psw[row * 36 + col]      = f2bf(p0);   // D-layout -> row-major staging
    psw[row * 36 + col + 16] = f2bf(p1);
  }
  asm volatile("s_wait_dscnt 0" ::: "memory");   // LDS RAW within wave
  v16bf ap = load_a_frag(psw + col * 36, 0);     // re-read as A-layout 16x32
  o0 = wmma_bf16(ap, load_bN_frag(vsb + 0,  72), o0);
  o1 = wmma_bf16(ap, load_bN_frag(vsb + 16, 72), o1);
  o2 = wmma_bf16(ap, load_bN_frag(vsb + 32, 72), o2);
  o3 = wmma_bf16(ap, load_bN_frag(vsb + 48, 72), o3);
}

// ---------- head 0: global attention (flash over 4096 keys) -----------------
// K/V chunks fetched by the TDM (contiguous 32x64 run, DMA pads each 128B
// token row to 144B); double-buffered against the WMMA/softmax work.

__global__ __launch_bounds__(128) void attn_global(
    const u16* __restrict__ q, const u16* __restrict__ k,
    const u16* __restrict__ v, u16* __restrict__ fused) {
  __shared__ u16 ks[2][32 * 72];
  __shared__ u16 vs[2][32 * 72];
  __shared__ u16 ps[4 * 16 * 36];
  const int b = blockIdx.x >> 6;            // 64 blocks per batch
  const int q0 = (blockIdx.x & 63) * 64;
  const int wv = threadIdx.x >> 5;
  const int lane = threadIdx.x & 31;
  const int hi = lane >> 4, col = lane & 15;
  const size_t hb = (size_t)b * NH * NTOK * HD;    // head 0
  const u16* qb_ = q + hb;
  const u16* kb_ = k + hb;
  const u16* vb_ = v + hb;
  const int m0 = q0 + wv * 16;
  const u16* qrow = qb_ + (size_t)(m0 + col) * HD;
  v16bf aq0 = load_a_frag(qrow, 0);
  v16bf aq1 = load_a_frag(qrow, 32);
  v8f o0 = {0,0,0,0,0,0,0,0}, o1 = o0, o2 = o0, o3 = o0;
  float m_r[8], l_r[8];
#pragma unroll
  for (int e = 0; e < 8; ++e) { m_r[e] = -1e30f; l_r[e] = 0.f; }
  u16* myps = ps + wv * 16 * 36;

  auto stage = [&](int kt, int buf) {
    if (threadIdx.x < 32) {
      const size_t go = (size_t)kt * 32 * HD;
      tdm_load_2d(lds_addr_of(&ks[buf][0]), kb_ + go,
                  2048u, 1u, 2048u, 2048u, 1u, 4u, 3u);
      tdm_load_2d(lds_addr_of(&vs[buf][0]), vb_ + go,
                  2048u, 1u, 2048u, 2048u, 1u, 4u, 3u);
    }
  };

  stage(0, 0);
  if (threadIdx.x < 32) __builtin_amdgcn_s_wait_tensorcnt(0);
  __syncthreads();
  for (int kt = 0; kt < NTOK / 32; ++kt) {
    const int cur = kt & 1;
    if (kt + 1 < NTOK / 32) stage(kt + 1, cur ^ 1);
    v8f s0 = {0,0,0,0,0,0,0,0}, s1 = s0;
    s0 = wmma_bf16(aq0, load_bT_frag(&ks[cur][0], 72), s0);
    s0 = wmma_bf16(aq1, load_bT_frag(&ks[cur][32], 72), s0);
    s1 = wmma_bf16(aq0, load_bT_frag(&ks[cur][16 * 72], 72), s1);
    s1 = wmma_bf16(aq1, load_bT_frag(&ks[cur][16 * 72 + 32], 72), s1);
    softmax_pv_step(s0, s1, m_r, l_r, o0, o1, o2, o3, myps, vs[cur]);
    if (threadIdx.x < 32) __builtin_amdgcn_s_wait_tensorcnt(0);
    __syncthreads();
  }

#pragma unroll
  for (int e = 0; e < 8; ++e) {
    float inv = 1.0f / l_r[e];
    size_t base = ((size_t)b * NTOK + (m0 + e + (hi << 3))) * CDIM;  // head 0
    fused[base + col]      = f2bf(o0[e] * inv);
    fused[base + col + 16] = f2bf(o1[e] * inv);
    fused[base + col + 32] = f2bf(o2[e] * inv);
    fused[base + col + 48] = f2bf(o3[e] * inv);
  }
}

// ---------- heads 1..7: windowed attention with relative-position bias ------
// A key chunk is min(32,Twin)/p window-rows of p*64 contiguous bf16 with
// stride 64*64 -> a 2D TDM tile, DMA-padded to the 144B token stride.

__global__ __launch_bounds__(512) void attn_window(
    const u16* __restrict__ q, const u16* __restrict__ k,
    const u16* __restrict__ v, const float* __restrict__ table,
    u16* __restrict__ fused, int p, int plog2, int head) {
  __shared__ u16 ks[2][32 * 72];
  __shared__ u16 vs[2][32 * 72];
  __shared__ u16 ps[16 * 16 * 36];
  __shared__ float tb[1024];
  const int wcols = WGRID >> plog2;
  const int nwin = wcols * wcols;
  const int b = blockIdx.x / nwin;
  const int win = blockIdx.x % nwin;
  const int wy = win / wcols, wx = win % wcols;
  const int Twin = p * p;
  const int wv = threadIdx.x >> 5;
  const int lane = threadIdx.x & 31;
  const int hi = lane >> 4, col = lane & 15;
  const int pm1 = p - 1, dpm1 = 2 * p - 1;
  const int tsz = dpm1 * dpm1;
  for (int i = threadIdx.x; i < tsz; i += blockDim.x) tb[i] = table[i];
  if (Twin < 32) {   // zero rows 16..31 so masked P columns hit zeros, not junk
    for (int i = threadIdx.x; i < 16 * 36; i += blockDim.x) {
      ((unsigned*)&ks[0][16 * 72])[i] = 0u;
      ((unsigned*)&vs[0][16 * 72])[i] = 0u;
    }
  }

  const size_t hb = ((size_t)b * NH + head) * NTOK * HD;
  const u16* qb_ = q + hb;
  const u16* kb_ = k + hb;
  const u16* vb_ = v + hb;

  int qt0 = wv * 16 + col;
  int qtok = ((wy * p + (qt0 >> plog2)) << 6) + wx * p + (qt0 & pm1);
  const u16* qrow = qb_ + (size_t)qtok * HD;
  v16bf aq0 = load_a_frag(qrow, 0);
  v16bf aq1 = load_a_frag(qrow, 32);
  v8f o0 = {0,0,0,0,0,0,0,0}, o1 = o0, o2 = o0, o3 = o0;
  float m_r[8], l_r[8];
#pragma unroll
  for (int e = 0; e < 8; ++e) { m_r[e] = -1e30f; l_r[e] = 0.f; }
  u16* myps = ps + wv * 16 * 36;
  const int nch = (Twin + 31) >> 5;
  const int cw = Twin < 32 ? Twin : 32;        // keys per chunk
  const unsigned xrows = (unsigned)(cw >> plog2);
  const unsigned run = (unsigned)(p * HD);     // contiguous elems per win-row

  auto stagew = [&](int kt, int buf) {
    if (threadIdx.x < 32) {
      int iy0 = (kt * 32) >> plog2;
      int tok0 = ((wy * p + iy0) << 6) + wx * p;
      tdm_load_2d(lds_addr_of(&ks[buf][0]), kb_ + (size_t)tok0 * HD,
                  run, 0x10000u, (unsigned)(WGRID * HD), run, xrows, 4u, 3u);
      tdm_load_2d(lds_addr_of(&vs[buf][0]), vb_ + (size_t)tok0 * HD,
                  run, 0x10000u, (unsigned)(WGRID * HD), run, xrows, 4u, 3u);
    }
  };

  stagew(0, 0);
  if (threadIdx.x < 32) __builtin_amdgcn_s_wait_tensorcnt(0);
  __syncthreads();
  for (int kt = 0; kt < nch; ++kt) {
    const int cur = kt & 1;
    if (kt + 1 < nch) stagew(kt + 1, cur ^ 1);
    v8f s0 = {0,0,0,0,0,0,0,0}, s1 = s0;
    s0 = wmma_bf16(aq0, load_bT_frag(&ks[cur][0], 72), s0);
    s0 = wmma_bf16(aq1, load_bT_frag(&ks[cur][32], 72), s0);
    s1 = wmma_bf16(aq0, load_bT_frag(&ks[cur][16 * 72], 72), s1);
    s1 = wmma_bf16(aq1, load_bT_frag(&ks[cur][16 * 72 + 32], 72), s1);
    const int key0 = kt * 32 + col, key1 = key0 + 16;
#pragma unroll
    for (int e = 0; e < 8; ++e) {
      int qr = wv * 16 + e + (hi << 3);
      int qy = qr >> plog2, qx = qr & pm1;
      if (key0 < Twin)
        s0[e] += tb[(qy - (key0 >> plog2) + pm1) * dpm1 + (qx - (key0 & pm1) + pm1)];
      else
        s0[e] = -1e30f;
      if (key1 < Twin)
        s1[e] += tb[(qy - (key1 >> plog2) + pm1) * dpm1 + (qx - (key1 & pm1) + pm1)];
      else
        s1[e] = -1e30f;
    }
    softmax_pv_step(s0, s1, m_r, l_r, o0, o1, o2, o3, myps, vs[cur]);
    if (threadIdx.x < 32) __builtin_amdgcn_s_wait_tensorcnt(0);
    __syncthreads();
  }

#pragma unroll
  for (int e = 0; e < 8; ++e) {
    float inv = 1.0f / l_r[e];
    int qr = wv * 16 + e + (hi << 3);
    int tok = ((wy * p + (qr >> plog2)) << 6) + wx * p + (qr & pm1);
    size_t base = ((size_t)b * NTOK + tok) * CDIM + (size_t)head * HD;
    fused[base + col]      = f2bf(o0[e] * inv);
    fused[base + col + 16] = f2bf(o1[e] * inv);
    fused[base + col + 32] = f2bf(o2[e] * inv);
    fused[base + col + 48] = f2bf(o3[e] * inv);
  }
}

// ---------- launch ----------------------------------------------------------

extern "C" void kernel_launch(void* const* d_in, const int* in_sizes, int n_in,
                              void* d_out, int out_size, void* d_ws, size_t ws_size,
                              hipStream_t stream) {
  (void)in_sizes; (void)n_in; (void)out_size; (void)ws_size;
  const float* x   = (const float*)d_in[0];
  const float* Wq  = (const float*)d_in[1];
  const float* Wkv = (const float*)d_in[2];
  const float* Wp  = (const float*)d_in[3];
  const float* bp  = (const float*)d_in[4];
  const float* tabs[7];
  for (int i = 0; i < 7; ++i) tabs[i] = (const float*)d_in[5 + i];
  float* out = (float*)d_out;

  const size_t nx = (size_t)BATCH * NTOK * CDIM;   // 16.78M elements
  char* w = (char*)d_ws;
  u16* xb   = (u16*)w; w += nx * 2;
  u16* wqb  = (u16*)w; w += (size_t)CDIM * CDIM * 2;
  u16* wkvb = (u16*)w; w += (size_t)2 * CDIM * CDIM * 2;
  u16* wpb  = (u16*)w; w += (size_t)CDIM * CDIM * 2;
  u16* qb   = (u16*)w; w += nx * 2;
  u16* kb   = (u16*)w; w += nx * 2;
  u16* vb   = (u16*)w; w += nx * 2;
  u16* fb   = (u16*)w; w += nx * 2;

  cvt_bf16<<<dim3(2048), dim3(256), 0, stream>>>(x, xb, (int)nx);
  cvt_bf16<<<dim3(128), dim3(256), 0, stream>>>(Wq, wqb, CDIM * CDIM);
  cvt_bf16<<<dim3(256), dim3(256), 0, stream>>>(Wkv, wkvb, 2 * CDIM * CDIM);
  cvt_bf16<<<dim3(128), dim3(256), 0, stream>>>(Wp, wpb, CDIM * CDIM);

  const int M = BATCH * NTOK;
  // q = x @ Wq^T, pre-scaled by hd^-0.5 = 0.125, scattered to (B,NH,N,HD)
  gemm_bf16<<<dim3(M / 128, CDIM / 64), 256, 0, stream>>>(
      xb, wqb, M, CDIM, CDIM, nullptr, nullptr, qb, nullptr, 1, 0.125f);
  // kv = x @ Wkv^T, split-scattered to k / v
  gemm_bf16<<<dim3(M / 128, 2 * CDIM / 64), 256, 0, stream>>>(
      xb, wkvb, M, 2 * CDIM, CDIM, nullptr, nullptr, kb, vb, 2, 1.0f);

  attn_global<<<dim3(BATCH * (NTOK / 64)), 128, 0, stream>>>(qb, kb, vb, fb);

  const int heads_p[7]  = {4, 4, 8, 8, 8, 16, 16};
  const int heads_lg[7] = {2, 2, 3, 3, 3, 4, 4};
  for (int h = 1; h < 8; ++h) {
    int p = heads_p[h - 1], lg = heads_lg[h - 1];
    int wcols = WGRID >> lg;
    int nwin = wcols * wcols;
    int threads = (p * p / 16) * 32;
    attn_window<<<dim3(BATCH * nwin), threads, 0, stream>>>(
        qb, kb, vb, tabs[h - 1], fb, p, lg, h);
  }

  // out = fused @ Wproj^T + bproj  (f32)
  gemm_bf16<<<dim3(M / 128, CDIM / 64), 256, 0, stream>>>(
      fb, wpb, M, CDIM, CDIM, out, bp, nullptr, nullptr, 0, 1.0f);
}
